// _pExpert_644245095187
// MI455X (gfx1250) — matrix operand order repaired and checked by reading, not compiled, not run on hardware
//
#include <hip/hip_runtime.h>

// ---------------------------------------------------------------------------
// MoE expert FFN for gfx1250 (MI455X, wave32, WMMA + TDM).
//   GEMM1: h = gelu(x @ w1[e])   (A fp32 -> bf16 on the fly, out bf16 to ws)
//   GEMM2: out = h @ w2[e]       (A bf16 staged via tensor_load_to_lds)
// bf16 operands, fp32 accumulate via v_wmma_f32_16x16x32_bf16.
// ---------------------------------------------------------------------------

typedef __attribute__((ext_vector_type(16))) __bf16 v16bf;
typedef __attribute__((ext_vector_type(8)))  float  v8f;
typedef unsigned int v4u  __attribute__((ext_vector_type(4)));
typedef int          v8it __attribute__((ext_vector_type(8)));
typedef int          v4it __attribute__((ext_vector_type(4)));

#define TILE_M 128
#define TILE_N 128
#define TILE_K 32
#define LDA    40   // padded LDS row stride (bf16 elems): conflict-free b128 reads
#define LDB    40

#if __has_builtin(__builtin_amdgcn_tensor_load_to_lds) && \
    __has_builtin(__builtin_amdgcn_s_wait_tensorcnt)
#define HAS_TDM 1
#else
#define HAS_TDM 0
#endif

// hardware RTNE f32 -> bf16 (let the backend pick v_cvt forms)
__device__ __forceinline__ unsigned short bf16_bits(float f) {
  union { __bf16 h; unsigned short u; } c;
  c.h = (__bf16)f;
  return c.u;
}
__device__ __forceinline__ unsigned pack_bf16(float lo, float hi) {
  union { __bf16 h[2]; unsigned u; } c;
  c.h[0] = (__bf16)lo; c.h[1] = (__bf16)hi;
  return c.u;
}

// tanh-approx gelu using TRANS ops (exp2 + rcp) that co-issue with WMMA
__device__ __forceinline__ float gelu_tanh(float x) {
  const float c0 = 0.7978845608028654f;   // sqrt(2/pi)
  const float c1 = 0.044715f;
  float y  = c0 * (x + c1 * x * x * x);
  float ex = __builtin_amdgcn_exp2f(y * 2.8853900817779268f);  // exp(2y)
  float th = 1.0f - 2.0f * __builtin_amdgcn_rcpf(ex + 1.0f);   // tanh(y)
  return 0.5f * x * (1.0f + th);
}

#if HAS_TDM
// Issue one TDM transfer: bf16 tile [TILE_M rows x TILE_K cols] from a
// row-major [*, kdim] bf16 matrix into LDS with 16-byte padding after each
// 64-byte row (LDS row stride = 80B = LDA bf16 elems).
__device__ __forceinline__ void tdm_load_a_tile(const unsigned short* gsrc,
                                                unsigned ldsOff, int kdim) {
  unsigned long long ga = (unsigned long long)(uintptr_t)gsrc;
  v4u g0;
  g0.x = 0x1u;                                   // count=1, user mode
  g0.y = ldsOff;                                 // lds_addr (bytes)
  g0.z = (unsigned)ga;                           // global_addr[31:0]
  g0.w = (unsigned)((ga >> 32) & 0x01FFFFFFull)  // global_addr[56:32]
         | 0x80000000u;                          // type=2 ("image")
  v8it g1 = {};
  // data_size=1 (2B) | pad_enable | pad_interval=3 (16 DW) | pad_amount=3 (4 DW)
  g1[0] = (1 << 16) | (1 << 20) | (3 << 22) | (3 << 25);
  g1[1] = (kdim & 0xFFFF) << 16;                 // tensor_dim0[15:0]   @ bits 63:48
  g1[2] = (kdim >> 16) | (TILE_M << 16);         // tensor_dim0[31:16] | tensor_dim1[15:0]
  g1[3] = (TILE_K << 16);                        // tensor_dim1[31:16]=0 | tile_dim0
  g1[4] = TILE_M;                                // tile_dim1 | tile_dim2=0
  g1[5] = kdim;                                  // tensor_dim0_stride[31:0]
  g1[6] = 0;
  g1[7] = 0;
  v4it z4 = {};
#if __has_include(<hip/amd_detail/amd_gfx1250_TDM.h>)
  v8it z8 = {};
  __builtin_amdgcn_tensor_load_to_lds(g0, g1, z4, z4, z8, 0);   // clang-23 form
#else
  __builtin_amdgcn_tensor_load_to_lds(g0, g1, z4, z4, 0);       // ROCm 7.2 form
#endif
}
#endif

template<int KDIM, int NDIM, bool A_IS_BF16, bool GELU_BF16_OUT>
__global__ __launch_bounds__(256)
void moe_wmma_gemm(const void* __restrict__ Av,
                   const float* __restrict__ W,      // [E][KDIM][NDIM] fp32
                   const int* __restrict__ counts,   // [E]
                   int nExperts,
                   void* __restrict__ Outv)
{
  __shared__ unsigned short sA[2][TILE_M * LDA];
  __shared__ unsigned short sB[2][TILE_N * LDB];

  const int t  = threadIdx.x;
  const int n0 = blockIdx.x * TILE_N;
  const int m0 = blockIdx.y * TILE_M;

  // expert of this token tile (token tiles assumed expert-aligned)
  int e = 0, cum = 0;
  for (int i = 0; i < nExperts; ++i) { cum += counts[i]; if (m0 < cum) { e = i; break; } }
  const float* We = W + (size_t)e * KDIM * NDIM;

  // wave / lane geometry (wave32)
  const int wave  = t >> 5;
  const int lane  = t & 31;
  const int lrow  = lane & 15;
  const int lhalf = lane >> 4;
  const int waveM = (wave & 1) * 64;   // 2 waves along M (4 m-tiles each)
  const int waveN = (wave >> 1) * 32;  // 4 waves along N (2 n-tiles each)

  // staging work assignment: A tile 128x32, B tile 32x128, 16 elems/thread
  const int am = t >> 1;              // A row 0..127
  const int ak = (t & 1) * 16;        // A k-offset 0 or 16
  const int bk = t >> 3;              // B k-row 0..31
  const int bn = (t & 7) * 16;        // B n-offset 0..112

#if HAS_TDM
  constexpr bool USE_TDM = A_IS_BF16;   // bf16 A needs no conversion -> TDM path
#else
  constexpr bool USE_TDM = false;
#endif

  v8f acc[4][2];
#pragma unroll
  for (int i = 0; i < 4; ++i)
#pragma unroll
    for (int j = 0; j < 2; ++j)
      acc[i][j] = (v8f){0.f, 0.f, 0.f, 0.f, 0.f, 0.f, 0.f, 0.f};

  // register staging for global->LDS double buffering (VGPR path)
  float    fa[16];
  unsigned uA[8];
  float    fb[16];

  auto loadA = [&](int kk) {
    if constexpr (A_IS_BF16) {
      const uint4* p = (const uint4*)((const unsigned short*)Av +
                                      (size_t)(m0 + am) * KDIM + kk + ak);
      uint4 q0 = p[0], q1 = p[1];
      uA[0] = q0.x; uA[1] = q0.y; uA[2] = q0.z; uA[3] = q0.w;
      uA[4] = q1.x; uA[5] = q1.y; uA[6] = q1.z; uA[7] = q1.w;
    } else {
      const float4* p = (const float4*)((const float*)Av +
                                        (size_t)(m0 + am) * KDIM + kk + ak);
      float4 v0 = p[0], v1 = p[1], v2 = p[2], v3 = p[3];
      fa[0]=v0.x; fa[1]=v0.y; fa[2]=v0.z; fa[3]=v0.w;
      fa[4]=v1.x; fa[5]=v1.y; fa[6]=v1.z; fa[7]=v1.w;
      fa[8]=v2.x; fa[9]=v2.y; fa[10]=v2.z; fa[11]=v2.w;
      fa[12]=v3.x; fa[13]=v3.y; fa[14]=v3.z; fa[15]=v3.w;
    }
  };

  auto storeA = [&](int buf) {
    unsigned short* d = &sA[buf][am * LDA + ak];
    unsigned pk[8];
    if constexpr (A_IS_BF16) {
#pragma unroll
      for (int i = 0; i < 8; ++i) pk[i] = uA[i];
    } else {
#pragma unroll
      for (int i = 0; i < 8; ++i) pk[i] = pack_bf16(fa[2 * i], fa[2 * i + 1]);
    }
    ((uint4*)d)[0] = make_uint4(pk[0], pk[1], pk[2], pk[3]);
    ((uint4*)d)[1] = make_uint4(pk[4], pk[5], pk[6], pk[7]);
  };

  auto loadB = [&](int kk) {
    const float4* p = (const float4*)(We + (size_t)(kk + bk) * NDIM + n0 + bn);
    float4 v0 = p[0], v1 = p[1], v2 = p[2], v3 = p[3];
    fb[0]=v0.x; fb[1]=v0.y; fb[2]=v0.z; fb[3]=v0.w;
    fb[4]=v1.x; fb[5]=v1.y; fb[6]=v1.z; fb[7]=v1.w;
    fb[8]=v2.x; fb[9]=v2.y; fb[10]=v2.z; fb[11]=v2.w;
    fb[12]=v3.x; fb[13]=v3.y; fb[14]=v3.z; fb[15]=v3.w;
    // L2 latency hiding for the fp32 weight panel two stages ahead
    if (kk + 2 * TILE_K < KDIM)
      __builtin_prefetch(We + (size_t)(kk + 2 * TILE_K + bk) * NDIM + n0 + bn, 0, 1);
  };

  auto storeB = [&](int buf) {   // transpose into [n][k] layout
#pragma unroll
    for (int i = 0; i < 16; ++i)
      sB[buf][(bn + i) * LDB + bk] = bf16_bits(fb[i]);
  };

  auto compute = [&](int buf) {
    union { v16bf v; uint4 q[2]; } afr[4], bfr[2];
#pragma unroll
    for (int mt = 0; mt < 4; ++mt) {
      // A frag: lane holds row (lrow), K = lhalf*8..+7 and 16+lhalf*8..+7
      const unsigned short* p = &sA[buf][(waveM + mt * 16 + lrow) * LDA + lhalf * 8];
      afr[mt].q[0] = *(const uint4*)p;
      afr[mt].q[1] = *(const uint4*)(p + 16);
    }
#pragma unroll
    for (int nt = 0; nt < 2; ++nt) {
      // B frag: lane holds col (lrow), K = lhalf*8..+7 and 16+lhalf*8..+7 (in [n][k])
      const unsigned short* p = &sB[buf][(waveN + nt * 16 + lrow) * LDB + lhalf * 8];
      bfr[nt].q[0] = *(const uint4*)p;
      bfr[nt].q[1] = *(const uint4*)(p + 16);
    }
#pragma unroll
    for (int mt = 0; mt < 4; ++mt)
#pragma unroll
      for (int nt = 0; nt < 2; ++nt)
        acc[mt][nt] = __builtin_amdgcn_wmma_f32_16x16x32_bf16(
            false, afr[mt].v, false, bfr[nt].v,
            (short)0, acc[mt][nt], false, false);
  };

#if HAS_TDM
  auto tdmA = [&](int kk, int buf) {
    if (t < 32) {  // wave 0 only (wave-uniform branch); TDM ignores EXEC lanes
      const unsigned short* g = (const unsigned short*)Av + (size_t)m0 * KDIM + kk;
      // generic-pointer low 32 bits == LDS byte offset (flat aperture rule)
      tdm_load_a_tile(g, (unsigned)(uintptr_t)&sA[buf][0], KDIM);
    }
  };
#endif

  // prologue: fill buffer 0
  if constexpr (USE_TDM) {
#if HAS_TDM
    tdmA(0, 0);
#endif
  } else {
    loadA(0);
  }
  loadB(0);
  if constexpr (!USE_TDM) storeA(0);
  storeB(0);

  const int nk = KDIM / TILE_K;
  for (int ki = 0; ki < nk; ++ki) {
    if constexpr (USE_TDM) {
#if HAS_TDM
      if (t < 32) __builtin_amdgcn_s_wait_tensorcnt(0);  // before barrier signal
#endif
    }
    __syncthreads();
    const int cur = ki & 1;
    if (ki + 1 < nk) {
      if constexpr (USE_TDM) {
#if HAS_TDM
        tdmA((ki + 1) * TILE_K, cur ^ 1);
#endif
      } else {
        loadA((ki + 1) * TILE_K);
      }
      loadB((ki + 1) * TILE_K);
    }
    compute(cur);
    if (ki + 1 < nk) {
      if constexpr (!USE_TDM) storeA(cur ^ 1);
      storeB(cur ^ 1);
    }
  }

  // epilogue: C/D layout — VGPR r, lanes 0-15: M=r, lanes 16-31: M=8+r; N=lane&15
#pragma unroll
  for (int mt = 0; mt < 4; ++mt) {
#pragma unroll
    for (int nt = 0; nt < 2; ++nt) {
#pragma unroll
      for (int r = 0; r < 8; ++r) {
        const int row = m0 + waveM + mt * 16 + lhalf * 8 + r;
        const int col = n0 + waveN + nt * 16 + lrow;
        const float v = acc[mt][nt][r];
        if constexpr (GELU_BF16_OUT) {
          ((unsigned short*)Outv)[(size_t)row * NDIM + col] = bf16_bits(gelu_tanh(v));
        } else {
          ((float*)Outv)[(size_t)row * NDIM + col] = v;
        }
      }
    }
  }
}

// ---------------------------------------------------------------------------

extern "C" void kernel_launch(void* const* d_in, const int* in_sizes, int n_in,
                              void* d_out, int out_size, void* d_ws, size_t ws_size,
                              hipStream_t stream) {
  constexpr int E = 8, D = 1024, H = 4096, T = 8192;

  const float* inp = (const float*)d_in[0];        // [T, D] fp32
  const float* w1  = (const float*)d_in[1];        // [E, D, H] fp32
  const float* w2  = (const float*)d_in[2];        // [E, H, D] fp32
  const int*   cnt = (const int*)d_in[3];          // [E] int32
  float*       out = (float*)d_out;                // [T, D] fp32

  unsigned short* h_bf16 = (unsigned short*)d_ws;  // [T, H] bf16 intermediate

  dim3 blk(256);

  // GEMM1: h = gelu(x @ w1[e])  -> bf16 workspace
  dim3 g1(H / TILE_N, T / TILE_M);
  moe_wmma_gemm<D, H, /*A bf16*/ false, /*gelu+bf16 out*/ true>
      <<<g1, blk, 0, stream>>>(inp, w1, cnt, E, h_bf16);

  // GEMM2: out = h @ w2[e]      -> fp32 output
  dim3 g2(D / TILE_N, T / TILE_M);
  moe_wmma_gemm<H, D, /*A bf16*/ true, /*gelu+bf16 out*/ false>
      <<<g2, blk, 0, stream>>>(h_bf16, w2, cnt, E, out);
}
